// Cache_64707977282190
// MI455X (gfx1250) — matrix operand order repaired and checked by reading, not compiled, not run on hardware
//
#include <hip/hip_runtime.h>
#include <math.h>

#define N_CACHE 262144
#define D       512
#define VOCAB   50257
#define INV_SMOOTH 5.0f   // 1 / 0.2

typedef __attribute__((ext_vector_type(16))) _Float16 v16h;
typedef __attribute__((ext_vector_type(8)))  _Float16 v8h;
typedef __attribute__((ext_vector_type(8)))  float    v8f;
typedef __attribute__((ext_vector_type(4)))  unsigned int u32x4;
typedef __attribute__((ext_vector_type(8)))  int          i32x8;
typedef __attribute__((ext_vector_type(4)))  int          i32x4;

union V16 { v16h v; v8h h[2]; };

__global__ __launch_bounds__(256) void zero_bins(float* __restrict__ bins, int n) {
  int i = blockIdx.x * blockDim.x + threadIdx.x;
  if (i < n) bins[i] = 0.0f;
}

// Issue one TDM tensor load: 128 rows x 32 cols of f32 (tile of cache_h) -> LDS.
// D# group0: count=1, lds_addr, global_addr (57b), type=2 ("image").
// D# group1: data_size=4B, tensor dims set huge (no OOB clipping),
//            tile_dim0=32, tile_dim1=128, tensor_dim0_stride=512.
__device__ __forceinline__ void tdm_load_tile(const float* gptr, unsigned lds_off) {
  unsigned long long ga = (unsigned long long)(uintptr_t)gptr;
  u32x4 g0;
  g0.x = 1u;                                              // count=1, user mode
  g0.y = lds_off;                                         // LDS byte address
  g0.z = (unsigned)ga;                                    // global_addr[31:0]
  g0.w = (unsigned)((ga >> 32) & 0x01FFFFFFu) | (2u << 30); // addr[56:32] | type=2
  i32x8 g1;
  g1[0] = 0x00020000;          // workgroup_mask=0, data_size=2 (4 bytes)
  g1[1] = 0;                   // tensor_dim0 lo16 = 0  (tensor_dim0 = 1<<20)
  g1[2] = 0x00000010;          // tensor_dim0 hi16 = 0x10, tensor_dim1 lo16 = 0
  g1[3] = (32 << 16) | 0x10;   // tile_dim0 = 32, tensor_dim1 hi16 = 0x10
  g1[4] = 128;                 // tile_dim1 = 128, tile_dim2 = 0
  g1[5] = 512;                 // tensor_dim0_stride = 512 (elements)
  g1[6] = 0;
  g1[7] = 0;
  i32x4 gz = {0, 0, 0, 0};
#if __clang_major__ >= 23
  i32x8 gz8 = {0, 0, 0, 0, 0, 0, 0, 0};
  __builtin_amdgcn_tensor_load_to_lds(g0, g1, gz, gz, gz8, 0);
#else
  __builtin_amdgcn_tensor_load_to_lds(g0, g1, gz, gz, 0);
#endif
}

// 8 waves / block, 16 rows / wave -> 128 cache rows per block.
// TDM double-buffers raw f32 128x32 tiles into LDS; waves build f16 diff
// fragments (A and B=E^T layouts) on the fly and accumulate diag(E*E^T)
// with v_wmma_f32_16x16x32_f16.
__global__ __launch_bounds__(256) void dist_scatter(
    const float* __restrict__ h_t,
    const float* __restrict__ cache_h,
    const int*   __restrict__ word_ids,
    float*       __restrict__ bins)
{
  __shared__ __align__(32) float hs[D];
  __shared__ __align__(32) float buf[2][128 * 32];   // double-buffered TDM tiles

  const int tid  = threadIdx.x;
  const int lane = tid & 31;
  const int wave = tid >> 5;

  hs[tid]       = h_t[tid];
  hs[tid + 256] = h_t[tid + 256];

  const int    rowBase = blockIdx.x * 128;
  const float* gbase   = cache_h + (size_t)rowBase * D;
  // low 32 bits of the generic (flat) address of a __shared__ object = LDS byte offset
  const unsigned lds0 = (unsigned)(uintptr_t)&buf[0][0];
  const unsigned lds1 = (unsigned)(uintptr_t)&buf[1][0];

  if (wave == 0) tdm_load_tile(gbase, lds0);           // prime buffer 0 (kk=0)
  __syncthreads();                                     // hs ready for everyone

  const int ar = lane & 15, hi = lane >> 4;
  v8f acc = {};

  const int tiles = D / 32;                            // 16
  for (int t = 0; t < tiles; ++t) {
    const int bi = t & 1;
    if (wave == 0) {
      if (t + 1 < tiles) {
        tdm_load_tile(gbase + (size_t)(t + 1) * 32, (bi ? lds0 : lds1));
        __builtin_amdgcn_s_wait_tensorcnt(1);          // tile t done, prefetch in flight
      } else {
        __builtin_amdgcn_s_wait_tensorcnt(0);
      }
    }
    __syncthreads();                                   // tile t visible to all waves

    const float* rowp = &buf[bi][(wave * 16 + ar) * 32];
    const float* hp   = &hs[t * 32];

    // A (16x32 f16): lanes 0-15 -> K 0..7 & 16..23 of row ar; lanes 16-31 -> K 8..15 & 24..31
    v8f ra0 = *(const v8f*)(rowp + hi * 8);
    v8f ra1 = *(const v8f*)(rowp + 16 + hi * 8);
    v8f ha0 = *(const v8f*)(hp + hi * 8);
    v8f ha1 = *(const v8f*)(hp + 16 + hi * 8);
    V16 a;
    #pragma unroll
    for (int j = 0; j < 8; ++j) {
      a.v[j]     = (_Float16)(ra0[j] - ha0[j]);
      a.v[j + 8] = (_Float16)(ra1[j] - ha1[j]);
    }
    // B (32x16 f16) = E^T: lanes 0-15 -> K 0..7 & 16..23 of col ar; lanes 16-31 -> K 8..15 & 24..31
    v8f rb0 = *(const v8f*)(rowp + hi * 8);
    v8f rb1 = *(const v8f*)(rowp + 16 + hi * 8);
    v8f hb0 = *(const v8f*)(hp + hi * 8);
    v8f hb1 = *(const v8f*)(hp + 16 + hi * 8);
    V16 b;
    #pragma unroll
    for (int j = 0; j < 8; ++j) {
      b.v[j]     = (_Float16)(rb0[j] - hb0[j]);
      b.v[j + 8] = (_Float16)(rb1[j] - hb1[j]);
    }

    acc = __builtin_amdgcn_wmma_f32_16x16x32_f16(
        false, a.v, false, b.v, (short)0, acc, false, false);

    __syncthreads();   // compute on tile t done before buffer bi is re-filled at t+2
  }

  // diagonal extraction: VGPR r = M=r (lanes 0-15) / M=r+8 (lanes 16-31), N=lane%16
  int m, ai; bool own;
  if (lane < 8)        { m = lane;      ai = lane;      own = true;  }
  else if (lane >= 24) { m = lane - 16; ai = lane - 24; own = true;  }
  else                 { m = 0;         ai = 0;         own = false; }
  float d2 = acc[0];
  #pragma unroll
  for (int i = 1; i < 8; ++i) d2 = (ai == i) ? acc[i] : d2;

  if (own) {
    const int row = rowBase + wave * 16 + m;
    float dist = sqrtf(fmaxf(d2, 0.0f));
    float kern = expf(dist * INV_SMOOTH);
    atomicAdd(&bins[word_ids[row]], kern);
  }
}

__global__ __launch_bounds__(1024) void log_softmax_k(const float* __restrict__ bins,
                                                      float* __restrict__ out)
{
  __shared__ float red[1024];
  __shared__ float s_max, s_logZ;
  const int tid = threadIdx.x;

  float lmax = -INFINITY;
  for (int i = tid; i < VOCAB; i += 1024) lmax = fmaxf(lmax, bins[i]);
  red[tid] = lmax; __syncthreads();
  for (int s = 512; s > 0; s >>= 1) {
    if (tid < s) red[tid] = fmaxf(red[tid], red[tid + s]);
    __syncthreads();
  }
  if (tid == 0) s_max = red[0];
  __syncthreads();
  const float mx = s_max;

  float lsum = 0.0f;
  for (int i = tid; i < VOCAB; i += 1024) lsum += expf(bins[i] - mx);
  red[tid] = lsum; __syncthreads();
  for (int s = 512; s > 0; s >>= 1) {
    if (tid < s) red[tid] += red[tid + s];
    __syncthreads();
  }
  if (tid == 0) s_logZ = mx + logf(red[0]);
  __syncthreads();
  const float lz = s_logZ;

  for (int i = tid; i < VOCAB; i += 1024) out[i] = bins[i] - lz;
}

extern "C" void kernel_launch(void* const* d_in, const int* in_sizes, int n_in,
                              void* d_out, int out_size, void* d_ws, size_t ws_size,
                              hipStream_t stream) {
  const float* h_t      = (const float*)d_in[0];   // [512] f32
  const float* cache_h  = (const float*)d_in[1];   // [262144, 512] f32
  const int*   word_ids = (const int*)d_in[2];     // [262144] i32
  float*       out      = (float*)d_out;           // [1, 50257] f32
  float*       bins     = (float*)d_ws;            // 50257 f32 scratch

  zero_bins<<<(VOCAB + 255) / 256, 256, 0, stream>>>(bins, VOCAB);
  dist_scatter<<<N_CACHE / 128, 256, 0, stream>>>(h_t, cache_h, word_ids, bins);
  log_softmax_k<<<1, 1024, 0, stream>>>(bins, out);
}
